// SAGE_13975823581723
// MI455X (gfx1250) — matrix operand order repaired and checked by reading, not compiled, not run on hardware
//
#include <hip/hip_runtime.h>

// ---------------------------------------------------------------------------
// GraphSAGE (pool aggregator) forward for MI455X / gfx1250.
// bf16 WMMA GEMMs (f32 accumulate), LDS-staged transposed weights,
// u32-atomic-max scatter (valid since messages are post-ReLU >= 0).
// ---------------------------------------------------------------------------

typedef __attribute__((ext_vector_type(16))) __bf16 bf16x16;
typedef __attribute__((ext_vector_type(8)))  float  f32x8;

union Frag32B { uint4 u[2]; bf16x16 v; };

__device__ __forceinline__ unsigned short f2bf(float f) {
    unsigned u = __float_as_uint(f);
    u += 0x7FFFu + ((u >> 16) & 1u);      // round-to-nearest-even
    return (unsigned short)(u >> 16);
}

// --------------------------- small utility kernels -------------------------

__global__ void k_f32_to_bf16(const float* __restrict__ in,
                              unsigned short* __restrict__ out, int n) {
    int i = blockIdx.x * 256 + threadIdx.x;
    if (i < n) out[i] = f2bf(in[i]);
}

__global__ void k_zero_f32(float* __restrict__ p, int n) {
    int i = blockIdx.x * 256 + threadIdx.x;
    if (i < n) p[i] = 0.0f;
}

// W [K, Ncols] f32 (row-major)  ->  WT [Ncols, K] bf16 (row-major)
__global__ void k_prep_wt(const float* __restrict__ W,
                          unsigned short* __restrict__ WT, int K, int Ncols) {
    int idx = blockIdx.x * 256 + threadIdx.x;
    if (idx >= K * Ncols) return;
    int n = idx / K, k = idx - n * K;
    WT[idx] = f2bf(W[k * Ncols + n]);
}

// Per-edge scatter-max of p[src] into pooled[dst]; p >= 0 so u32 bit-pattern
// max == float max, and 0-initialized pooled gives the isolated-node 0.
__global__ void k_scatter_max(const float* __restrict__ p,
                              const int* __restrict__ src,
                              const int* __restrict__ dst,
                              unsigned int* __restrict__ pooled, int E) {
    long long tid = (long long)blockIdx.x * 256 + threadIdx.x;
    if (tid >= (long long)E * 32) return;
    int e = (int)(tid >> 5);
    int c = ((int)tid & 31) << 2;               // 4-float chunk
    int s = src[e], d = dst[e];
    float4 v = *(const float4*)(p + (size_t)s * 128 + c);
    unsigned int* q = pooled + (size_t)d * 128 + c;
    atomicMax(q + 0, __float_as_uint(v.x));
    atomicMax(q + 1, __float_as_uint(v.y));
    atomicMax(q + 2, __float_as_uint(v.z));
    atomicMax(q + 3, __float_as_uint(v.w));
}

// ------------------------------ WMMA GEMM ----------------------------------
// out[M, NC] = act( A1[M,128] @ W1 + (DUAL ? A2[M,128] @ W2 : 0) + bias )
// A*: bf16 row-major.  W*: transposed bf16 [NC, 128] row-major.
// 256 threads = 8 waves; wave w owns rows [blk*128 + w*16, +16), all NC cols.
// act: 0 = none, 1 = relu, 2 = leaky_relu(0.01)
template <int NTILES, bool DUAL>
__global__ __launch_bounds__(256)
void k_gemm_wmma(const unsigned short* __restrict__ A1,
                 const unsigned short* __restrict__ W1,
                 const unsigned short* __restrict__ A2,
                 const unsigned short* __restrict__ W2,
                 const float* __restrict__ bias,
                 float* __restrict__ outF,
                 unsigned short* __restrict__ outB,
                 int M, int act) {
    constexpr int NC = NTILES * 16;
    constexpr int LS = 136;                     // padded LDS row stride (elems)
    __shared__ unsigned short sW1[NC * LS];
    __shared__ unsigned short sW2[DUAL ? NC * LS : 1];

    const int tid = threadIdx.x;
    // cooperative stage of W (NC rows x 128 bf16 = NC*16 uint4 chunks)
    for (int i = tid; i < NC * 16; i += 256) {
        int row = i >> 4, vtx = i & 15;
        *(uint4*)&sW1[row * LS + vtx * 8] = ((const uint4*)W1)[row * 16 + vtx];
        if (DUAL)
            *(uint4*)&sW2[row * LS + vtx * 8] = ((const uint4*)W2)[row * 16 + vtx];
    }
    __syncthreads();

    const int wave = tid >> 5, lane = tid & 31;
    const int hv = lane >> 4, lr = lane & 15;   // half-wave, lane-in-half
    const int m0 = blockIdx.x * 128 + wave * 16;
    int arow = m0 + lr;
    if (arow > M - 1) arow = M - 1;             // clamp (EXEC must stay full)

    f32x8 acc[NTILES];
#pragma unroll
    for (int t = 0; t < NTILES; ++t)
        acc[t] = (f32x8){0.f, 0.f, 0.f, 0.f, 0.f, 0.f, 0.f, 0.f};

    const unsigned short* a1p = A1 + (size_t)arow * 128;
    const unsigned short* a2p = DUAL ? (A2 + (size_t)arow * 128) : nullptr;

#pragma unroll
    for (int k0 = 0; k0 < 128; k0 += 32) {
        Frag32B a1;
        a1.u[0] = *(const uint4*)(a1p + k0 + hv * 16);
        a1.u[1] = *(const uint4*)(a1p + k0 + hv * 16 + 8);
#pragma unroll
        for (int t = 0; t < NTILES; ++t) {
            Frag32B b;
            const unsigned short* bp = &sW1[(t * 16 + lr) * LS + k0 + hv * 16];
            b.u[0] = *(const uint4*)(bp);
            b.u[1] = *(const uint4*)(bp + 8);
            acc[t] = __builtin_amdgcn_wmma_f32_16x16x32_bf16(
                false, a1.v, false, b.v, (short)0, acc[t], false, false);
        }
        if (DUAL) {
            Frag32B a2;
            a2.u[0] = *(const uint4*)(a2p + k0 + hv * 16);
            a2.u[1] = *(const uint4*)(a2p + k0 + hv * 16 + 8);
#pragma unroll
            for (int t = 0; t < NTILES; ++t) {
                Frag32B b;
                const unsigned short* bp = &sW2[(t * 16 + lr) * LS + k0 + hv * 16];
                b.u[0] = *(const uint4*)(bp);
                b.u[1] = *(const uint4*)(bp + 8);
                acc[t] = __builtin_amdgcn_wmma_f32_16x16x32_bf16(
                    false, a2.v, false, b.v, (short)0, acc[t], false, false);
            }
        }
    }

    // epilogue: C/D layout -> m = m0 + r + 8*half, n = t*16 + (lane&15)
#pragma unroll
    for (int t = 0; t < NTILES; ++t) {
        int n = t * 16 + lr;
        float bv = bias[n];
#pragma unroll
        for (int r = 0; r < 8; ++r) {
            int m = m0 + r + hv * 8;
            if (m < M) {
                float v = acc[t][r] + bv;
                if (act == 1) v = v > 0.f ? v : 0.f;
                else if (act == 2) v = v > 0.f ? v : 0.01f * v;
                if (outF) outF[(size_t)m * NC + n] = v;
                if (outB) outB[(size_t)m * NC + n] = f2bf(v);
            }
        }
    }
}

// ------------------------------- launcher ----------------------------------

extern "C" void kernel_launch(void* const* d_in, const int* in_sizes, int n_in,
                              void* d_out, int out_size, void* d_ws, size_t ws_size,
                              hipStream_t stream) {
    const float* x   = (const float*)d_in[0];
    const int*   src = (const int*)d_in[1];
    const int*   dst = (const int*)d_in[2];
    const float* Wp1 = (const float*)d_in[3];  const float* bp1 = (const float*)d_in[4];
    const float* Ws1 = (const float*)d_in[5];  const float* Wn1 = (const float*)d_in[6];
    const float* bc1 = (const float*)d_in[7];
    const float* Wp2 = (const float*)d_in[8];  const float* bp2 = (const float*)d_in[9];
    const float* Ws2 = (const float*)d_in[10]; const float* Wn2 = (const float*)d_in[11];
    const float* bc2 = (const float*)d_in[12];
    const float* Wl[5] = {(const float*)d_in[13], (const float*)d_in[15],
                          (const float*)d_in[17], (const float*)d_in[19],
                          (const float*)d_in[21]};
    const float* bl[5] = {(const float*)d_in[14], (const float*)d_in[16],
                          (const float*)d_in[18], (const float*)d_in[20],
                          (const float*)d_in[22]};

    const int N = in_sizes[0] / 128;
    const int E = in_sizes[1];
    const size_t NF = (size_t)N * 128;

    // workspace carve-up (256B aligned chunks)
    char* w = (char*)d_ws;
    auto take = [&](size_t bytes) {
        char* p = w;
        w += (bytes + 255) & ~(size_t)255;
        return p;
    };
    unsigned short* xb  = (unsigned short*)take(NF * 2);
    unsigned short* hA  = (unsigned short*)take(NF * 2);
    unsigned short* hB  = (unsigned short*)take(NF * 2);
    unsigned short* plb = (unsigned short*)take(NF * 2);
    float* pbuf   = (float*)take(NF * 4);
    float* pooled = (float*)take(NF * 4);
    unsigned short* WT[11];
    for (int i = 0; i < 11; ++i) WT[i] = (unsigned short*)take(128 * 128 * 2);
    // WT index map: 0=Wp1 1=Ws1 2=Wn1 3=Wp2 4=Ws2 5=Wn2 6..9=Wl1..4 10=Wl5

    const int TPB = 256;
    const int gElem = (int)((NF + TPB - 1) / TPB);
    const int gW128 = (128 * 128 + TPB - 1) / TPB;
    const int gW16  = (128 * 16 + TPB - 1) / TPB;
    const int gGemm = (N + 127) / 128;
    const long long eWork = (long long)E * 32;
    const int gEdge = (int)((eWork + TPB - 1) / TPB);

    // ---- prep: bf16 input + transposed bf16 weights ----
    hipLaunchKernelGGL(k_f32_to_bf16, dim3(gElem), dim3(TPB), 0, stream, x, xb, (int)NF);
    const float* Wsrc[11] = {Wp1, Ws1, Wn1, Wp2, Ws2, Wn2, Wl[0], Wl[1], Wl[2], Wl[3], Wl[4]};
    for (int i = 0; i < 10; ++i)
        hipLaunchKernelGGL(k_prep_wt, dim3(gW128), dim3(TPB), 0, stream, Wsrc[i], WT[i], 128, 128);
    hipLaunchKernelGGL(k_prep_wt, dim3(gW16), dim3(TPB), 0, stream, Wsrc[10], WT[10], 128, 16);

    // ---- conv1 ----
    // p = relu(x @ Wp1 + bp1)   (gather commuted out of fc_pool)
    hipLaunchKernelGGL((k_gemm_wmma<8, false>), dim3(gGemm), dim3(TPB), 0, stream,
                       xb, WT[0], (const unsigned short*)nullptr, (const unsigned short*)nullptr,
                       bp1, pbuf, (unsigned short*)nullptr, N, 1);
    hipLaunchKernelGGL(k_zero_f32, dim3(gElem), dim3(TPB), 0, stream, pooled, (int)NF);
    hipLaunchKernelGGL(k_scatter_max, dim3(gEdge), dim3(TPB), 0, stream,
                       pbuf, src, dst, (unsigned int*)pooled, E);
    hipLaunchKernelGGL(k_f32_to_bf16, dim3(gElem), dim3(TPB), 0, stream, pooled, plb, (int)NF);
    // hA = leaky(x @ Ws1 + pooled @ Wn1 + bc1)
    hipLaunchKernelGGL((k_gemm_wmma<8, true>), dim3(gGemm), dim3(TPB), 0, stream,
                       xb, WT[1], plb, WT[2], bc1, (float*)nullptr, hA, N, 2);

    // ---- conv2 ----
    hipLaunchKernelGGL((k_gemm_wmma<8, false>), dim3(gGemm), dim3(TPB), 0, stream,
                       hA, WT[3], (const unsigned short*)nullptr, (const unsigned short*)nullptr,
                       bp2, pbuf, (unsigned short*)nullptr, N, 1);
    hipLaunchKernelGGL(k_zero_f32, dim3(gElem), dim3(TPB), 0, stream, pooled, (int)NF);
    hipLaunchKernelGGL(k_scatter_max, dim3(gEdge), dim3(TPB), 0, stream,
                       pbuf, src, dst, (unsigned int*)pooled, E);
    hipLaunchKernelGGL(k_f32_to_bf16, dim3(gElem), dim3(TPB), 0, stream, pooled, plb, (int)NF);
    hipLaunchKernelGGL((k_gemm_wmma<8, true>), dim3(gGemm), dim3(TPB), 0, stream,
                       hA, WT[4], plb, WT[5], bc2, (float*)nullptr, hB, N, 2);

    // ---- dense stack: hB -> hA -> hB -> hA -> hB ----
    unsigned short* ping = hB;
    unsigned short* pong = hA;
    for (int l = 0; l < 4; ++l) {
        hipLaunchKernelGGL((k_gemm_wmma<8, false>), dim3(gGemm), dim3(TPB), 0, stream,
                           ping, WT[6 + l], (const unsigned short*)nullptr,
                           (const unsigned short*)nullptr,
                           bl[l], (float*)nullptr, pong, N, 2);
        unsigned short* t = ping; ping = pong; pong = t;
    }
    // final H -> C (16 cols), f32 out, no activation
    hipLaunchKernelGGL((k_gemm_wmma<1, false>), dim3(gGemm), dim3(TPB), 0, stream,
                       ping, WT[10], (const unsigned short*)nullptr,
                       (const unsigned short*)nullptr,
                       bl[4], (float*)d_out, (unsigned short*)nullptr, N, 0);
}
